// ShiftedSparseAttention_54133767799177
// MI455X (gfx1250) — matrix operand order repaired and checked by reading, not compiled, not run on hardware
//
#include <hip/hip_runtime.h>

// ---------------------------------------------------------------------------
// Shifted Sparse Attention for MI455X (gfx1250, wave32, WMMA).
// The reference's rolls are identity (shift -128 on a size-16 axis), so the
// math is: qkv = x @ W^T + b, then softmax attention inside independent
// 128-token blocks per head. GEMM (103 GFLOP) dominates -> f16 WMMA with f32
// accumulation; the whole working set (~250MB traffic) is L2-resident.
// ---------------------------------------------------------------------------

typedef __attribute__((ext_vector_type(16))) _Float16 v16h;
typedef __attribute__((ext_vector_type(8)))  _Float16 v8h;
typedef __attribute__((ext_vector_type(4)))  _Float16 v4h;
typedef __attribute__((ext_vector_type(8)))  float    v8f;

#define NB   2
#define SEQ  8192
#define HID  1024
#define NH   16
#define HD   64
#define NOUT 3072   // 3*HID

union U16h { v16h v; v8h h[2]; };

static __device__ __forceinline__ v8f wmma_f16(const v16h& a, const v16h& b, const v8f& c) {
    return __builtin_amdgcn_wmma_f32_16x16x32_f16(false, a, false, b, (short)0, c, false, false);
}

// --------------------------- W f32 -> f16 ----------------------------------
__global__ __launch_bounds__(256) void cvt_f32_f16(const float* __restrict__ src,
                                                   _Float16* __restrict__ dst, int n4) {
    int i = blockIdx.x * blockDim.x + threadIdx.x;
    if (i < n4) {
        float4 f = ((const float4*)src)[i];
        v4h o;
        o[0] = (_Float16)f.x; o[1] = (_Float16)f.y;
        o[2] = (_Float16)f.z; o[3] = (_Float16)f.w;
        ((v4h*)dst)[i] = o;
    }
}

// --------------------------- QKV projection --------------------------------
// Each wave computes a 32(M) x 64(N) tile of qkv = hidden @ W^T.
// A fragment (16x32 f16): lane holds row m=lane&15, 16 contiguous K at aklo.
// B fragment (32x16 f16): lane holds col n=lane&15, 16 contiguous K at bklo.
// Epilogue adds bias and scatters f16 into Q,K [b][h][seq][64] and V^T
// [b][h][64][seq] so the attention kernel's B-fragments are contiguous.
__global__ __launch_bounds__(256) void qkv_gemm(
    const float*    __restrict__ hidden,  // [NB*SEQ][HID] f32
    const _Float16* __restrict__ wh,      // [NOUT][HID]   f16
    const float*    __restrict__ bias,    // [NOUT]
    _Float16* __restrict__ qw,            // [NB][NH][SEQ][HD]
    _Float16* __restrict__ kw,            // [NB][NH][SEQ][HD]
    _Float16* __restrict__ vtw)           // [NB][NH][HD][SEQ]
{
    const int lane = threadIdx.x & 31;
    const int wave = (blockIdx.x * blockDim.x + threadIdx.x) >> 5;
    const int NN   = NOUT / 64;               // 48 N-tiles of 64
    const int tileM = (wave / NN) * 32;       // 512 M bands
    const int tileN = (wave % NN) * 64;

    const int lrow = lane & 15;
    const int hi   = lane >> 4;
    const int aklo = hi << 4;                 // 0 / 16
    const int bklo = hi << 4;                 // 0 / 16

    v8f acc[2][4];
#pragma unroll
    for (int mt = 0; mt < 2; ++mt)
#pragma unroll
        for (int nt = 0; nt < 4; ++nt) acc[mt][nt] = (v8f)(0.0f);

    for (int k0 = 0; k0 < HID; k0 += 32) {
        v16h a[2];
#pragma unroll
        for (int mt = 0; mt < 2; ++mt) {
            const float* p = hidden + (size_t)(tileM + mt * 16 + lrow) * HID + k0 + aklo;
            float4 f0 = *(const float4*)(p);
            float4 f1 = *(const float4*)(p + 4);
            float4 f2 = *(const float4*)(p + 8);
            float4 f3 = *(const float4*)(p + 12);
            v16h t;
            t[0]  = (_Float16)f0.x; t[1]  = (_Float16)f0.y; t[2]  = (_Float16)f0.z; t[3]  = (_Float16)f0.w;
            t[4]  = (_Float16)f1.x; t[5]  = (_Float16)f1.y; t[6]  = (_Float16)f1.z; t[7]  = (_Float16)f1.w;
            t[8]  = (_Float16)f2.x; t[9]  = (_Float16)f2.y; t[10] = (_Float16)f2.z; t[11] = (_Float16)f2.w;
            t[12] = (_Float16)f3.x; t[13] = (_Float16)f3.y; t[14] = (_Float16)f3.z; t[15] = (_Float16)f3.w;
            a[mt] = t;
        }
        v16h bfr[4];
#pragma unroll
        for (int nt = 0; nt < 4; ++nt)
            bfr[nt] = *(const v16h*)(wh + (size_t)(tileN + nt * 16 + lrow) * HID + k0 + bklo);
#pragma unroll
        for (int mt = 0; mt < 2; ++mt)
#pragma unroll
            for (int nt = 0; nt < 4; ++nt)
                acc[mt][nt] = wmma_f16(a[mt], bfr[nt], acc[mt][nt]);
    }

#pragma unroll
    for (int nt = 0; nt < 4; ++nt) {
        const int ng   = tileN + nt * 16 + lrow;   // output channel
        const float bv = bias[ng];
        const int sel  = ng >> 10;                  // 0=q 1=k 2=v
        const int rem  = ng & 1023;
        const int head = rem >> 6;
        const int dim  = rem & 63;
#pragma unroll
        for (int mt = 0; mt < 2; ++mt) {
#pragma unroll
            for (int r = 0; r < 8; ++r) {
                const int mg  = tileM + mt * 16 + r + (hi << 3);
                const int bb  = mg >> 13;
                const int seq = mg & 8191;
                const _Float16 hv = (_Float16)(acc[mt][nt][r] + bv);
                const size_t bh = (size_t)(bb * NH + head);
                if (sel == 0)      qw[(bh * SEQ + seq) * HD + dim] = hv;
                else if (sel == 1) kw[(bh * SEQ + seq) * HD + dim] = hv;
                else               vtw[(bh * HD + dim) * SEQ + seq] = hv;
            }
        }
    }
}

// --------------------------- Block attention -------------------------------
// One workgroup per (b, head, 128-token block); wave w owns queries
// [16w, 16w+16). S = Q K^T / 8 (16 WMMAs), softmax in registers with
// shfl_xor row reductions, P staged through padded LDS (D-layout -> A-layout
// reshape), O = P V (16 WMMAs) written f32 to d_out.
__global__ __launch_bounds__(256) void sattn(
    const _Float16* __restrict__ qw,
    const _Float16* __restrict__ kw,
    const _Float16* __restrict__ vtw,
    float* __restrict__ out)
{
    __shared__ _Float16 sw[8][16][136];   // +8 f16 pad: conflict-free b128 reads

    const int lane = threadIdx.x & 31;
    const int wave = threadIdx.x >> 5;
    const int blk  = blockIdx.x & 63;
    const int h    = (blockIdx.x >> 6) & 15;
    const int b    = blockIdx.x >> 10;

    const int lrow = lane & 15;
    const int hi   = lane >> 4;
    const int aklo = hi << 4;
    const int bklo = hi << 4;

    const size_t bh = (size_t)(b * NH + h);
    const _Float16* qp = qw  + (bh * SEQ + blk * 128 + wave * 16) * HD;
    const _Float16* kp = kw  + (bh * SEQ + blk * 128) * HD;
    const _Float16* vp = vtw + bh * HD * SEQ + blk * 128;

    // Q fragments: K = head_dim = 64 -> 2 k-steps.
    v16h qa[2];
#pragma unroll
    for (int ks = 0; ks < 2; ++ks) {
        const _Float16* p = qp + lrow * HD + ks * 32 + aklo;
        U16h u; u.h[0] = *(const v8h*)p; u.h[1] = *(const v8h*)(p + 8);
        qa[ks] = u.v;
    }

    // Scores S (16 x 128): 8 N-tiles over the block's keys.
    v8f st[8];
#pragma unroll
    for (int t = 0; t < 8; ++t) st[t] = (v8f)(0.0f);
#pragma unroll
    for (int t = 0; t < 8; ++t)
#pragma unroll
        for (int ks = 0; ks < 2; ++ks) {
            v16h kb = *(const v16h*)(kp + (size_t)(t * 16 + lrow) * HD + ks * 32 + bklo);
            st[t] = wmma_f16(qa[ks], kb, st[t]);
        }

    // Softmax over 128 keys per row. Row m = r + 8*hi lives in VGPR r across
    // 16 lanes of one half-wave -> shfl_xor masks 1,2,4,8 reduce exactly it.
    const float scale = 0.125f;  // 1/sqrt(64)
    float inv[8];
#pragma unroll
    for (int r = 0; r < 8; ++r) {
        float m = st[0][r];
#pragma unroll
        for (int t = 1; t < 8; ++t) m = fmaxf(m, st[t][r]);
        m = fmaxf(m, __shfl_xor(m, 1, 32));
        m = fmaxf(m, __shfl_xor(m, 2, 32));
        m = fmaxf(m, __shfl_xor(m, 4, 32));
        m = fmaxf(m, __shfl_xor(m, 8, 32));
        float s = 0.0f;
#pragma unroll
        for (int t = 0; t < 8; ++t) {
            float p = __expf((st[t][r] - m) * scale);
            st[t][r] = p;
            s += p;
        }
        s += __shfl_xor(s, 1, 32);
        s += __shfl_xor(s, 2, 32);
        s += __shfl_xor(s, 4, 32);
        s += __shfl_xor(s, 8, 32);
        inv[r] = 1.0f / s;
    }

    // D-layout -> LDS [m][n] as f16 probabilities.
#pragma unroll
    for (int t = 0; t < 8; ++t)
#pragma unroll
        for (int r = 0; r < 8; ++r)
            sw[wave][r + (hi << 3)][t * 16 + lrow] = (_Float16)(st[t][r] * inv[r]);
    __syncthreads();

    // A fragments of P (16 x 128): 4 k-steps over keys.
    v16h wa[4];
#pragma unroll
    for (int ks = 0; ks < 4; ++ks) {
        const _Float16* p = &sw[wave][lrow][ks * 32 + aklo];
        U16h u; u.h[0] = *(const v8h*)p; u.h[1] = *(const v8h*)(p + 8);
        wa[ks] = u.v;
    }

    // O = P @ V : N = 64 dims (4 tiles), K = 128 keys (4 steps); V^T is
    // contiguous along keys per lane.
    v8f ot[4];
#pragma unroll
    for (int nt = 0; nt < 4; ++nt) ot[nt] = (v8f)(0.0f);
#pragma unroll
    for (int nt = 0; nt < 4; ++nt)
#pragma unroll
        for (int ks = 0; ks < 4; ++ks) {
            v16h vb = *(const v16h*)(vp + (size_t)(nt * 16 + lrow) * SEQ + ks * 32 + bklo);
            ot[nt] = wmma_f16(wa[ks], vb, ot[nt]);
        }

    const size_t orow = (size_t)b * SEQ + blk * 128 + wave * 16;
#pragma unroll
    for (int nt = 0; nt < 4; ++nt)
#pragma unroll
        for (int r = 0; r < 8; ++r) {
            const int m = r + (hi << 3);
            out[(orow + m) * HID + h * HD + nt * 16 + lrow] = ot[nt][r];
        }
}

// ---------------------------------------------------------------------------
extern "C" void kernel_launch(void* const* d_in, const int* in_sizes, int n_in,
                              void* d_out, int out_size, void* d_ws, size_t ws_size,
                              hipStream_t stream) {
    const float* hidden = (const float*)d_in[0];   // [2,8192,1024]
    const float* w_qkv  = (const float*)d_in[1];   // [3072,1024]
    const float* b_qkv  = (const float*)d_in[2];   // [3072]
    float* out = (float*)d_out;                    // [2,8192,1024]

    // Workspace: W f16 (6.3MB) + Q,K,V^T f16 (3 x 33.6MB) ~= 102MB.
    _Float16* wh = (_Float16*)d_ws;
    size_t off = (size_t)NOUT * HID;
    _Float16* qw  = wh + off; off += (size_t)NB * NH * SEQ * HD;
    _Float16* kw  = wh + off; off += (size_t)NB * NH * SEQ * HD;
    _Float16* vtw = wh + off;

    const int n4 = (NOUT * HID) / 4;
    cvt_f32_f16<<<(n4 + 255) / 256, 256, 0, stream>>>(w_qkv, wh, n4);

    // 512 M-bands x 48 N-bands of wave tiles = 24576 waves = 3072 blocks.
    qkv_gemm<<<3072, 256, 0, stream>>>(hidden, wh, b_qkv, qw, kw, vtw);

    // 2 batch x 16 heads x 64 blocks of 128 tokens.
    sattn<<<NB * NH * (SEQ / 128), 256, 0, stream>>>(qw, kw, vtw, out);
}
